// SingleModelSelfAttn_62251255988298
// MI455X (gfx1250) — matrix operand order repaired and checked by reading, not compiled, not run on hardware
//
#include <hip/hip_runtime.h>
#include <hip/hip_bf16.h>
#include <cstdint>
#include <cstddef>

// ---------------------------------------------------------------------------
// MI455X (gfx1250) transformer encoder, compute path:
//   - every GEMM: v_wmma_f32_16x16x32_bf16 (bf16 in, f32 accumulate)
//   - all operands staged bf16; B always stored [N,K] so every LDS tile fill
//     is a contiguous 16B chunk -> async global->LDS copies (ASYNCcnt) with
//     double-buffered LDS to overlap copy of tile i+1 with WMMA on tile i
//   - 128x128 block tile (8 waves x 32x64, 16 WMMA / K-tile / wave);
//     64x64 variant for the N=64 attention-output GEMM
// ---------------------------------------------------------------------------

typedef __bf16 bf16;
typedef __attribute__((ext_vector_type(8)))  bf16  v8bf;
typedef __attribute__((ext_vector_type(16))) bf16  v16bf;
typedef __attribute__((ext_vector_type(8)))  float v8f;
typedef int v4i __attribute__((__vector_size__(4 * sizeof(int))));

#if __has_builtin(__builtin_amdgcn_global_load_async_to_lds_b128)
#define HAVE_ASYNC 1
#else
#define HAVE_ASYNC 0
#endif

__device__ __forceinline__ unsigned short f2bf_bits(float f) {
  union { float f; unsigned u; } un; un.f = f;
  unsigned u = un.u;
  u += 0x7FFFu + ((u >> 16) & 1u);          // round-to-nearest-even
  return (unsigned short)(u >> 16);
}
__device__ __forceinline__ bf16 f2bf(float f) {
  unsigned short s = f2bf_bits(f);
  return __builtin_bit_cast(bf16, s);
}

// 16B global -> LDS copy; async (ASYNCcnt-tracked) when the toolchain has it
__device__ __forceinline__ void cp16_g2l(const unsigned short* g, bf16* l) {
#if HAVE_ASYNC
  __builtin_amdgcn_global_load_async_to_lds_b128(
      (__attribute__((address_space(1))) v4i*)g,
      (__attribute__((address_space(3))) v4i*)l, 0, 0);
#else
  *(v8bf*)l = *(const v8bf*)g;
#endif
}

__device__ __forceinline__ void wait_async_copies() {
#if HAVE_ASYNC
#if __has_builtin(__builtin_amdgcn_s_wait_asynccnt)
  __builtin_amdgcn_s_wait_asynccnt(0);
#else
  asm volatile("s_wait_asynccnt 0x0" ::: "memory");
#endif
#endif
}

// ---------------------------------------------------------------------------
// WMMA GEMM:  C[M,N] = alpha * A[M,K] x B[N,K]^T + bias
//   A: bf16 [M,K] row-major, B: bf16 [N,K] row-major ("BT" always)
//   OUTMODE 0: f32 -> Cf[ldc]  | 1: bf16 -> Cb[ldc] | 2: bf16 transposed
//   (mode 2: token-row r, col c -> Cb[(batch(r)*512 + c)*S + r%S], ldc = S)
//   grid = (N/BN, M/BM), block = 256 threads (8 wave32s)
// ---------------------------------------------------------------------------
template <int BM, int BN, int WM, int WN, int OUTMODE, bool RELU>
__global__ __launch_bounds__(256)
void gemm_wmma(const unsigned short* __restrict__ A, int lda,
               const unsigned short* __restrict__ B, int ldb,
               float* __restrict__ Cf, unsigned short* __restrict__ Cb, int ldc,
               const float* __restrict__ bias, int K, float alpha) {
  constexpr int BK = 64;
  constexpr int FM = WM / 16;
  constexpr int FN = WN / 16;
  constexpr int WPM = BM / WM;                 // waves along M
  constexpr int ACH = BM * BK / (256 * 8);     // 16B chunks per thread (A)
  constexpr int BCH = BN * BK / (256 * 8);     // 16B chunks per thread (B)

  __shared__ bf16 As[2][BM][BK + 8];           // +8 keeps 16B row alignment
  __shared__ bf16 Bs[2][BN][BK + 8];

  const int t = threadIdx.x;
  const int rowBase = blockIdx.y * BM;
  const int colBase = blockIdx.x * BN;

  const int lane = t & 31;
  const int wid  = t >> 5;
  const int m0w  = (wid % WPM) * WM;
  const int n0w  = (wid / WPM) * WN;
  const int lh   = lane & 15;
  const int lg   = lane >> 4;                  // K-striping half-wave group

  v8f acc[FM][FN] = {};

  auto copy_tile = [&](int buf, int k0) {
#pragma unroll
    for (int i = 0; i < ACH; ++i) {
      const int g = (i * 256 + t) * 8;
      const int r = g >> 6, c = g & 63;
      cp16_g2l(A + (size_t)(rowBase + r) * lda + (k0 + c), &As[buf][r][c]);
    }
#pragma unroll
    for (int i = 0; i < BCH; ++i) {
      const int g = (i * 256 + t) * 8;
      const int r = g >> 6, c = g & 63;
      cp16_g2l(B + (size_t)(colBase + r) * ldb + (k0 + c), &Bs[buf][r][c]);
    }
  };

  const int ntiles = K >> 6;
  copy_tile(0, 0);

  for (int i = 0; i < ntiles; ++i) {
    wait_async_copies();
    __syncthreads();                           // tile i visible to all waves
    if (i + 1 < ntiles) copy_tile((i + 1) & 1, (i + 1) * BK);  // overlap
    const int buf = i & 1;

#pragma unroll
    for (int ks = 0; ks < BK; ks += 32) {
      v16bf afr[FM];
#pragma unroll
      for (int fm = 0; fm < FM; ++fm) {
        v8bf lo = *(const v8bf*)&As[buf][m0w + fm * 16 + lh][ks + lg * 8];
        v8bf hi = *(const v8bf*)&As[buf][m0w + fm * 16 + lh][ks + 16 + lg * 8];
        afr[fm] = __builtin_shufflevector(lo, hi, 0, 1, 2, 3, 4, 5, 6, 7,
                                          8, 9, 10, 11, 12, 13, 14, 15);
      }
      v16bf bfr[FN];
#pragma unroll
      for (int fn = 0; fn < FN; ++fn) {
        v8bf lo = *(const v8bf*)&Bs[buf][n0w + fn * 16 + lh][ks + lg * 8];
        v8bf hi = *(const v8bf*)&Bs[buf][n0w + fn * 16 + lh][ks + 16 + lg * 8];
        bfr[fn] = __builtin_shufflevector(lo, hi, 0, 1, 2, 3, 4, 5, 6, 7,
                                          8, 9, 10, 11, 12, 13, 14, 15);
      }
#pragma unroll
      for (int fm = 0; fm < FM; ++fm)
#pragma unroll
        for (int fn = 0; fn < FN; ++fn)
          acc[fm][fn] = __builtin_amdgcn_wmma_f32_16x16x32_bf16(
              false, afr[fm], false, bfr[fn], (short)0, acc[fm][fn],
              false, false);
    }
  }

  // Epilogue. C frag layout: VGPR r -> row r (lanes 0-15) / r+8 (lanes 16-31)
#pragma unroll
  for (int fm = 0; fm < FM; ++fm) {
#pragma unroll
    for (int fn = 0; fn < FN; ++fn) {
      const int row = rowBase + m0w + fm * 16 + lg * 8;
      const int col = colBase + n0w + fn * 16 + lh;
      const float bb = bias ? bias[col] : 0.f;
      if (OUTMODE == 2) {
        const int Svt = ldc;                 // mode 2: ldc carries S
        const int b   = row / Svt;
        const size_t base = ((size_t)(b * 512 + col)) * Svt + (row - b * Svt);
        v8bf ov;
#pragma unroll
        for (int r = 0; r < 8; ++r) {
          float v = acc[fm][fn][r] * alpha + bb;
          if (RELU) v = fmaxf(v, 0.f);
          ov[r] = f2bf(v);
        }
        *(v8bf*)&Cb[base] = ov;              // 8 consecutive tokens: one b128
      } else {
#pragma unroll
        for (int r = 0; r < 8; ++r) {
          float v = acc[fm][fn][r] * alpha + bb;
          if (RELU) v = fmaxf(v, 0.f);
          const size_t o = (size_t)(row + r) * ldc + col;
          if (OUTMODE == 1) Cb[o] = f2bf_bits(v);
          else              Cf[o] = v;
        }
      }
    }
  }
}

// ---------------------------------------------------------------------------
// Weight packing (all weights stored transposed: [N_out, K_in] bf16)
// ---------------------------------------------------------------------------
// Wq/Wk/Wv [H=8, D=512, A=64] f32 -> [HA=512, D=512] bf16 : out[(h*64+a)][d]
__global__ void pack_qkv_t(const float* __restrict__ in, unsigned short* __restrict__ out) {
  const int i = blockIdx.x * 256 + threadIdx.x;   // i = n*512 + d
  const int n = i >> 9, d = i & 511;
  const int h = n >> 6, a = n & 63;
  out[i] = f2bf_bits(in[h * 32768 + d * 64 + a]);
}
// square [512,512] transpose: out[n*512+k] = in[k*512+n]
__global__ void pack_t512(const float* __restrict__ in, unsigned short* __restrict__ out) {
  const int i = blockIdx.x * 256 + threadIdx.x;
  const int n = i >> 9, k = i & 511;
  out[i] = f2bf_bits(in[k * 512 + n]);
}
// Wp [512,256] -> [256,512]: out[t*512+d] = in[d*256+t]
__global__ void pack_wp_t(const float* __restrict__ in, unsigned short* __restrict__ out) {
  const int i = blockIdx.x * 256 + threadIdx.x;
  const int n = i >> 9, d = i & 511;
  out[i] = f2bf_bits(in[d * 256 + n]);
}
__global__ void conv_bf16(const float* __restrict__ in, unsigned short* __restrict__ out, int n) {
  const int i = blockIdx.x * 256 + threadIdx.x;
  if (i < n) out[i] = f2bf_bits(in[i]);
}

// ---------------------------------------------------------------------------
// Row softmax: f32 scores in, bf16 probabilities out (scale applied in GEMM)
// ---------------------------------------------------------------------------
__global__ __launch_bounds__(256)
void softmax_rows(float* __restrict__ sc, unsigned short* __restrict__ out, int S) {
  float* p = sc + (size_t)blockIdx.x * S;
  unsigned short* o = out + (size_t)blockIdx.x * S;
  __shared__ float red[256];
  const int t = threadIdx.x;
  float mx = -3.4e38f;
  for (int c = t; c < S; c += 256) mx = fmaxf(mx, p[c]);
  red[t] = mx; __syncthreads();
  for (int s = 128; s > 0; s >>= 1) { if (t < s) red[t] = fmaxf(red[t], red[t + s]); __syncthreads(); }
  const float m = red[0]; __syncthreads();
  float sum = 0.f;
  for (int c = t; c < S; c += 256) { float e = __expf(p[c] - m); p[c] = e; sum += e; }
  red[t] = sum; __syncthreads();
  for (int s = 128; s > 0; s >>= 1) { if (t < s) red[t] += red[t + s]; __syncthreads(); }
  const float inv = 1.f / red[0];
  for (int c = t; c < S; c += 256) o[c] = f2bf_bits(p[c] * inv);
}

// ---------------------------------------------------------------------------
// Fused residual + LayerNorm (D=512): x = LN(x + y)*g + b ; also bf16 mirror
// ---------------------------------------------------------------------------
__global__ __launch_bounds__(256)
void add_ln(float* __restrict__ x, unsigned short* __restrict__ xb,
            const float* __restrict__ y,
            const float* __restrict__ g, const float* __restrict__ b) {
  const int row = blockIdx.x;
  float* xp = x + (size_t)row * 512;
  unsigned short* xo = xb + (size_t)row * 512;
  const float* yp = y + (size_t)row * 512;
  const int t = threadIdx.x;
  float v0 = xp[t] + yp[t];
  float v1 = xp[t + 256] + yp[t + 256];
  __shared__ float red[256];
  red[t] = v0 + v1; __syncthreads();
  for (int s = 128; s > 0; s >>= 1) { if (t < s) red[t] += red[t + s]; __syncthreads(); }
  const float mean = red[0] * (1.f / 512.f); __syncthreads();
  const float d0 = v0 - mean, d1 = v1 - mean;
  red[t] = d0 * d0 + d1 * d1; __syncthreads();
  for (int s = 128; s > 0; s >>= 1) { if (t < s) red[t] += red[t + s]; __syncthreads(); }
  const float inv = rsqrtf(red[0] * (1.f / 512.f) + 1e-5f);
  const float r0 = d0 * inv * g[t] + b[t];
  const float r1 = d1 * inv * g[t + 256] + b[t + 256];
  xp[t] = r0;            xo[t] = f2bf_bits(r0);
  xp[t + 256] = r1;      xo[t + 256] = f2bf_bits(r1);
}

// ---------------------------------------------------------------------------
// Host-side dispatch
// ---------------------------------------------------------------------------
template <int OUTMODE, bool RELU>
static void gemm_big(hipStream_t st, int M, int N, int K,
                     const unsigned short* A, int lda,
                     const unsigned short* B, int ldb,
                     float* Cf, unsigned short* Cb, int ldc,
                     const float* bias, float alpha) {
  dim3 grid((unsigned)(N / 128), (unsigned)(M / 128));
  gemm_wmma<128, 128, 32, 64, OUTMODE, RELU>
      <<<grid, 256, 0, st>>>(A, lda, B, ldb, Cf, Cb, ldc, bias, K, alpha);
}
static void gemm_small_bf16(hipStream_t st, int M, int N, int K,
                            const unsigned short* A, int lda,
                            const unsigned short* B, int ldb,
                            unsigned short* Cb, int ldc, float alpha) {
  dim3 grid((unsigned)(N / 64), (unsigned)(M / 64));
  gemm_wmma<64, 64, 16, 32, 1, false>
      <<<grid, 256, 0, st>>>(A, lda, B, ldb, nullptr, Cb, ldc, nullptr, K, alpha);
}

extern "C" void kernel_launch(void* const* d_in, const int* in_sizes, int n_in,
                              void* d_out, int out_size, void* d_ws, size_t ws_size,
                              hipStream_t stream) {
  (void)in_sizes; (void)n_in; (void)out_size; (void)ws_size;
  constexpr int L = 2, H = 8, D = 512, A = 64, T = 256, BB = 2, S = 2048;
  constexpr int N  = BB * S;    // 4096 token rows
  constexpr int HA = H * A;     // 512
  constexpr size_t DD = (size_t)D * D;

  const float* x_in  = (const float*)d_in[0];
  const float* Wq    = (const float*)d_in[1];
  const float* bq    = (const float*)d_in[2];
  const float* Wk    = (const float*)d_in[3];
  const float* bk    = (const float*)d_in[4];
  const float* Wv    = (const float*)d_in[5];
  const float* bv    = (const float*)d_in[6];
  const float* Wo    = (const float*)d_in[7];
  const float* bo    = (const float*)d_in[8];
  const float* ln1_g = (const float*)d_in[9];
  const float* ln1_b = (const float*)d_in[10];
  const float* W1    = (const float*)d_in[11];
  const float* b1    = (const float*)d_in[12];
  const float* W2    = (const float*)d_in[13];
  const float* b2    = (const float*)d_in[14];
  const float* ln2_g = (const float*)d_in[15];
  const float* ln2_b = (const float*)d_in[16];
  const float* Wp    = (const float*)d_in[17];
  const float* bp    = (const float*)d_in[18];

  uint8_t* w = (uint8_t*)d_ws;
  auto carve = [&](size_t bytes) -> void* {
    void* p = (void*)w;
    w += (bytes + 255) & ~(size_t)255;
    return p;
  };
  float*          X   = (float*)carve((size_t)N * D * 4);           // residual
  unsigned short* Xb  = (unsigned short*)carve((size_t)N * D * 2);  // bf16 x
  unsigned short* Qb  = (unsigned short*)carve((size_t)N * D * 2);
  unsigned short* Kb  = (unsigned short*)carve((size_t)N * D * 2);
  unsigned short* Vt  = (unsigned short*)carve((size_t)N * D * 2);  // [B*H*A,S]
  unsigned short* HDb = (unsigned short*)carve((size_t)N * D * 2);  // heads
  float*          AT  = (float*)carve((size_t)N * D * 4);           // pre-LN
  unsigned short* F1b = (unsigned short*)carve((size_t)N * D * 2);  // relu(ffn)
  float*          SC  = (float*)carve((size_t)S * S * 4);           // scores
  unsigned short* SCb = (unsigned short*)carve((size_t)S * S * 2);  // probs
  unsigned short* WQB = (unsigned short*)carve((size_t)L * DD * 2);
  unsigned short* WKB = (unsigned short*)carve((size_t)L * DD * 2);
  unsigned short* WVB = (unsigned short*)carve((size_t)L * DD * 2);
  unsigned short* WOB = (unsigned short*)carve((size_t)L * DD * 2);
  unsigned short* W1B = (unsigned short*)carve((size_t)L * DD * 2);
  unsigned short* W2B = (unsigned short*)carve((size_t)L * DD * 2);
  unsigned short* WPB = (unsigned short*)carve((size_t)D * T * 2);

  // --- pack weights (transposed, bf16) ---
  const int nDD = (int)DD;  // 262144
  for (int l = 0; l < L; ++l) {
    pack_qkv_t<<<nDD / 256, 256, 0, stream>>>(Wq + (size_t)l * H * D * A, WQB + (size_t)l * DD);
    pack_qkv_t<<<nDD / 256, 256, 0, stream>>>(Wk + (size_t)l * H * D * A, WKB + (size_t)l * DD);
    pack_qkv_t<<<nDD / 256, 256, 0, stream>>>(Wv + (size_t)l * H * D * A, WVB + (size_t)l * DD);
    pack_t512 <<<nDD / 256, 256, 0, stream>>>(Wo + (size_t)l * DD, WOB + (size_t)l * DD);
    pack_t512 <<<nDD / 256, 256, 0, stream>>>(W1 + (size_t)l * DD, W1B + (size_t)l * DD);
    pack_t512 <<<nDD / 256, 256, 0, stream>>>(W2 + (size_t)l * DD, W2B + (size_t)l * DD);
  }
  pack_wp_t<<<(D * T) / 256, 256, 0, stream>>>(Wp, WPB);

  (void)hipMemcpyAsync(X, x_in, (size_t)N * D * sizeof(float),
                       hipMemcpyDeviceToDevice, stream);
  conv_bf16<<<(N * D) / 256, 256, 0, stream>>>(x_in, Xb, N * D);

  const float scale = 0.125f;  // 1/sqrt(A=64)

  for (int l = 0; l < L; ++l) {
    const unsigned short* wq = WQB + (size_t)l * DD;
    const unsigned short* wk = WKB + (size_t)l * DD;
    const unsigned short* wv = WVB + (size_t)l * DD;
    const unsigned short* wo = WOB + (size_t)l * DD;
    const unsigned short* w1 = W1B + (size_t)l * DD;
    const unsigned short* w2 = W2B + (size_t)l * DD;

    // QKV projections: [4096,512] x [512,512]^T
    gemm_big<1, false>(stream, N, HA, D, Xb, D, wq, D, nullptr, Qb, HA,
                       bq + (size_t)l * HA, 1.f);
    gemm_big<1, false>(stream, N, HA, D, Xb, D, wk, D, nullptr, Kb, HA,
                       bk + (size_t)l * HA, 1.f);
    gemm_big<2, false>(stream, N, HA, D, Xb, D, wv, D, nullptr, Vt, S,
                       bv + (size_t)l * HA, 1.f);   // writes V^T [B*H*A, S]

    for (int b = 0; b < BB; ++b) {
      for (int h = 0; h < H; ++h) {
        const unsigned short* Aq = Qb + (size_t)b * S * D + (size_t)h * A;
        const unsigned short* Bk = Kb + (size_t)b * S * D + (size_t)h * A;
        const unsigned short* Bv = Vt + ((size_t)(b * H + h) * A) * S;
        unsigned short*       Ch = HDb + (size_t)b * S * D + (size_t)h * A;
        // scores = scale * q @ k^T   [S,S]
        gemm_big<0, false>(stream, S, S, A, Aq, D, Bk, D, SC, nullptr, S,
                           nullptr, scale);
        softmax_rows<<<S, 256, 0, stream>>>(SC, SCb, S);
        // heads = attn @ v  (B = V^T stored [A,S])
        gemm_small_bf16(stream, S, A, S, SCb, S, Bv, S, Ch, D, 1.f);
      }
    }

    // output projection + AddNorm
    gemm_big<0, false>(stream, N, D, HA, HDb, D, wo, D, AT, nullptr, D,
                       bo + (size_t)l * D, 1.f);
    add_ln<<<N, 256, 0, stream>>>(X, Xb, AT, ln1_g + (size_t)l * D, ln1_b + (size_t)l * D);

    // FFN + AddNorm
    gemm_big<1, true >(stream, N, D, D, Xb, D, w1, D, nullptr, F1b, D,
                       b1 + (size_t)l * D, 1.f);
    gemm_big<0, false>(stream, N, D, D, F1b, D, w2, D, AT, nullptr, D,
                       b2 + (size_t)l * D, 1.f);
    add_ln<<<N, 256, 0, stream>>>(X, Xb, AT, ln2_g + (size_t)l * D, ln2_b + (size_t)l * D);
  }

  // final projection to target dim
  gemm_big<0, false>(stream, N, T, D, Xb, D, WPB, D, (float*)d_out, nullptr, T,
                     bp, 1.f);
}
